// MinGRU_10247791968754
// MI455X (gfx1250) — matrix operand order repaired and checked by reading, not compiled, not run on hardware
//
#include <hip/hip_runtime.h>
#include <math.h>

typedef __attribute__((ext_vector_type(8)))  __bf16 v8bf;
typedef __attribute__((ext_vector_type(16))) __bf16 v16bf;
typedef __attribute__((ext_vector_type(8)))  float  v8f;

#define B_  4
#define D_  512
#define L_  8192
#define O_  512

static __device__ __forceinline__ unsigned short f2bf(float f) {
  union { float f; unsigned u; } v; v.f = f;
  unsigned r = v.u + 0x7FFFu + ((v.u >> 16) & 1u);   // round-to-nearest-even
  return (unsigned short)(r >> 16);
}
static __device__ __forceinline__ float bf2f(unsigned short h) {
  union { unsigned u; float f; } v; v.u = ((unsigned)h) << 16;
  return v.f;
}

// Load a v16bf WMMA fragment from two aligned 16B halves at p and p+gap.
static __device__ __forceinline__ v16bf ldfrag(const unsigned short* p, int gap) {
  v8bf lo = *(const v8bf*)p;
  v8bf hi = *(const v8bf*)(p + gap);
  return __builtin_shufflevector(lo, hi,
      0, 1, 2, 3, 4, 5, 6, 7, 8, 9, 10, 11, 12, 13, 14, 15);
}

// ---------------------------------------------------------------------------
// Kernel 1: magnitude-preserving normalization of h_w (512x512) and g_w
// (510x512) rows -> bf16 weight matrices. Wg rows 0,1 are zero (the +-1000
// polarize rows are handled analytically in the scan kernel).
// ---------------------------------------------------------------------------
__global__ void prep_weights(const float* __restrict__ h_w,
                             const float* __restrict__ g_w,
                             unsigned short* __restrict__ Wh,
                             unsigned short* __restrict__ Wg) {
  __shared__ float red[256];
  int row = blockIdx.x;
  int t   = threadIdx.x;
  const float*    src;
  unsigned short* dst;
  if (row < O_) {
    src = h_w + (size_t)row * D_;
    dst = Wh  + (size_t)row * D_;
  } else {
    int o = row - O_;
    dst = Wg + (size_t)o * D_;
    if (o < 2) { dst[t] = 0; dst[t + 256] = 0; return; }
    src = g_w + (size_t)(o - 2) * D_;
  }
  float a = src[t], c = src[t + 256];
  red[t] = a * a + c * c;
  __syncthreads();
  for (int s = 128; s > 0; s >>= 1) {
    if (t < s) red[t] += red[t + s];
    __syncthreads();
  }
  float ss    = red[0];
  float sfan  = sqrtf((float)D_);
  float scale = 1.0f / ((1e-4f + sqrtf(ss) / sfan) * sfan);
  dst[t]       = f2bf(a * scale);
  dst[t + 256] = f2bf(c * scale);
}

// ---------------------------------------------------------------------------
// Kernel 2: depthwise 5-tap conv, pad=2, per-channel unit-norm weights.
// Output c stored bf16 in [B][L][D] layout (channel contiguous) so the GEMM
// B-operand loads are contiguous b128 loads.
// ---------------------------------------------------------------------------
__global__ void dwconv(const float* __restrict__ x,
                       const float* __restrict__ conv_w,
                       unsigned short* __restrict__ C) {
  __shared__ float xs[32][132];
  __shared__ float wn[32][5];
  int l0 = blockIdx.x * 128;
  int d0 = blockIdx.y * 32;
  int b  = blockIdx.z;
  int t  = threadIdx.x;

  for (int idx = t; idx < 32 * 132; idx += 256) {
    int dj = idx / 132, p = idx % 132;
    int gl = l0 - 2 + p;
    float v = 0.f;
    if (gl >= 0 && gl < L_) v = x[((size_t)(b * D_ + d0 + dj)) * L_ + gl];
    xs[dj][p] = v;
  }
  if (t < 32) {
    float w[5], ss = 0.f;
    #pragma unroll
    for (int k = 0; k < 5; k++) { w[k] = conv_w[(d0 + t) * 5 + k]; ss += w[k] * w[k]; }
    float s5 = sqrtf(5.f);
    float scale = 1.0f / ((1e-4f + sqrtf(ss) / s5) * s5);
    #pragma unroll
    for (int k = 0; k < 5; k++) wn[t][k] = w[k] * scale;
  }
  __syncthreads();

  for (int idx = t; idx < 32 * 128; idx += 256) {
    int dj = idx & 31, li = idx >> 5;   // consecutive threads -> consecutive d (coalesced store)
    float acc = 0.f;
    #pragma unroll
    for (int k = 0; k < 5; k++) acc += xs[dj][li + k] * wn[dj][k];
    C[((size_t)(b * L_ + l0 + li)) * D_ + d0 + dj] = f2bf(acc);
  }
}

// ---------------------------------------------------------------------------
// Kernel 3: batched GEMM via v_wmma_f32_16x16x32_bf16.
//   h[b] = Wh (512x512) x c[b] (512x8192), same for g.
// Block: 256 threads = 8 waves as 2(M) x 4(N) -> block tile 64(M) x 128(N).
// Wave tile: 32(M) x 32(N) = four 16x16 f32 accumulators; per K-step (K=32):
// 2 A-frags + 2 B-frags (8 x b128 loads) feed 4 WMMAs -> 1 WMMA / 2 loads.
// Operand layouts used here (contiguous K order, same for A and B; not the ISA table order):
//   A 16x32 bf16: lanes 0-15 row M=lane, elems K 0-15;
//                 lanes 16-31 same rows,  elems K 16-31.
//   B 32x16 bf16: lanes 0-15 col N=lane, K 0-15; lanes 16-31 K 16-31.
//   C/D f32: lane -> N=lane%16, vgpr j -> M = j (+8 for upper lanes).
// ---------------------------------------------------------------------------
__global__ void gemm_wmma(const unsigned short* __restrict__ Wh,
                          const unsigned short* __restrict__ Wg,
                          const unsigned short* __restrict__ C,
                          unsigned short* __restrict__ H,
                          unsigned short* __restrict__ G) {
  int z   = blockIdx.z;            // mat*4 + b
  int mat = z >> 2, b = z & 3;
  const unsigned short* W   = mat ? Wg : Wh;
  unsigned short*       OUT = mat ? G  : H;

  int lane = threadIdx.x & 31;
  int wave = threadIdx.x >> 5;
  int wm   = wave & 1;             // 2 M-waves
  int wn   = wave >> 1;            // 4 N-waves
  int m0   = blockIdx.y * 64  + wm * 32;
  int n0   = blockIdx.x * 128 + wn * 32;
  int l15  = lane & 15;
  bool lo16 = (lane < 16);

  v8f acc00 = {}; v8f acc01 = {}; v8f acc10 = {}; v8f acc11 = {};

  const unsigned short* A0 = W + (size_t)(m0 + l15) * D_ + (lo16 ? 0 : 16);
  const unsigned short* A1 = A0 + (size_t)16 * D_;
  const unsigned short* B0 = C + ((size_t)(b * L_ + n0 + l15)) * D_ + (lo16 ? 0 : 16);
  const unsigned short* B1 = B0 + (size_t)16 * D_;

  #pragma unroll 4
  for (int k0 = 0; k0 < D_; k0 += 32) {
    v16bf av0 = ldfrag(A0 + k0, 8);
    v16bf av1 = ldfrag(A1 + k0, 8);
    v16bf bv0 = ldfrag(B0 + k0, 8);
    v16bf bv1 = ldfrag(B1 + k0, 8);
    acc00 = __builtin_amdgcn_wmma_f32_16x16x32_bf16(
        false, av0, false, bv0, (short)0, acc00, false, false);
    acc01 = __builtin_amdgcn_wmma_f32_16x16x32_bf16(
        false, av0, false, bv1, (short)0, acc01, false, false);
    acc10 = __builtin_amdgcn_wmma_f32_16x16x32_bf16(
        false, av1, false, bv0, (short)0, acc10, false, false);
    acc11 = __builtin_amdgcn_wmma_f32_16x16x32_bf16(
        false, av1, false, bv1, (short)0, acc11, false, false);
  }

  int nOut  = n0 + l15;
  int mrow0 = m0 + (lo16 ? 0 : 8);
  #pragma unroll
  for (int j = 0; j < 8; j++) {
    size_t r0 = ((size_t)(b * O_ + mrow0 + j)) * L_;
    size_t r1 = ((size_t)(b * O_ + mrow0 + 16 + j)) * L_;
    OUT[r0 + nOut]      = f2bf(acc00[j]);
    OUT[r0 + nOut + 16] = f2bf(acc01[j]);
    OUT[r1 + nOut]      = f2bf(acc10[j]);
    OUT[r1 + nOut + 16] = f2bf(acc11[j]);
  }
}

// ---------------------------------------------------------------------------
// Kernel 4: minGRU linear recurrence o[t]=(1-s)*o[t-1]+s*h[t], s=sigmoid(g).
// One block per (b,o) row. Rows staged into LDS with CDNA5 async-to-LDS
// (GLOBAL_LOAD_ASYNC_TO_LDS_B128, tracked by ASYNCcnt); each wave waits its
// own asynccnt==0 before the workgroup barrier, so all LDS writes are visible
// to all consumers after the barrier. Then: chunked affine composition (A,B)
// per thread over 32 steps, Hillis-Steele scan of 256 pairs, replay.
// Rows o=0/1 use g=-1000/+1000 (sigmoid saturates exactly to 0/1 in fp32).
// ---------------------------------------------------------------------------
__global__ void scan_gru(const unsigned short* __restrict__ H,
                         const unsigned short* __restrict__ G,
                         float* __restrict__ out) {
  __shared__ unsigned short hs[L_];
  __shared__ unsigned short gs[L_];
  __shared__ float sA[256];
  __shared__ float sB[256];
  int o = blockIdx.x, b = blockIdx.y;
  int t = threadIdx.x;
  size_t rowoff = ((size_t)(b * O_ + o)) * L_;
  bool useconst = (o < 2);

  // async stage: 8192 bf16 = 1024 x 16B chunks per row
  unsigned hs_base = (unsigned)(unsigned long long)&hs[0];
  unsigned gs_base = (unsigned)(unsigned long long)&gs[0];
  unsigned long long hrow = (unsigned long long)(const void*)(H + rowoff);
  for (int i = t; i < L_ / 8; i += 256) {
    unsigned long long ga = hrow + (unsigned long long)i * 16u;
    unsigned la = hs_base + (unsigned)i * 16u;
    asm volatile("global_load_async_to_lds_b128 %0, %1, off"
                 :: "v"(la), "v"(ga) : "memory");
  }
  if (!useconst) {
    unsigned long long grow = (unsigned long long)(const void*)(G + rowoff);
    for (int i = t; i < L_ / 8; i += 256) {
      unsigned long long ga = grow + (unsigned long long)i * 16u;
      unsigned la = gs_base + (unsigned)i * 16u;
      asm volatile("global_load_async_to_lds_b128 %0, %1, off"
                   :: "v"(la), "v"(ga) : "memory");
    }
  }
  asm volatile("s_wait_asynccnt 0" ::: "memory");
  __syncthreads();

  float gconst = (o == 0) ? -1000.f : 1000.f;
  int base = t * 32;

  // pass 1: compose this thread's 32-step chunk into affine (A, B)
  float A = 1.f, Bc = 0.f;
  for (int i = 0; i < 32; i++) {
    float g = useconst ? gconst : bf2f(gs[base + i]);
    float s = 1.f / (1.f + __expf(-g));
    float h = bf2f(hs[base + i]);
    float a = 1.f - s;
    A  = a * A;
    Bc = a * Bc + s * h;
  }
  sA[t] = A; sB[t] = Bc;
  __syncthreads();

  // inclusive scan over 256 affine pairs: (A2,B2) o (A1,B1) = (A2*A1, A2*B1+B2)
  for (int off = 1; off < 256; off <<= 1) {
    float pa = 1.f, pb = 0.f;
    if (t >= off) { pa = sA[t - off]; pb = sB[t - off]; }
    float ca = sA[t], cb = sB[t];
    __syncthreads();
    sA[t] = ca * pa;
    sB[t] = ca * pb + cb;
    __syncthreads();
  }
  float hin = (t == 0) ? 0.f : sB[t - 1];   // h0 = 0

  // pass 2: replay chunk with carry-in and write outputs
  float ov = hin;
  for (int i = 0; i < 32; i++) {
    float g = useconst ? gconst : bf2f(gs[base + i]);
    float s = 1.f / (1.f + __expf(-g));
    float h = bf2f(hs[base + i]);
    ov = (1.f - s) * ov + s * h;
    out[rowoff + base + i] = ov;
  }
}

// ---------------------------------------------------------------------------
extern "C" void kernel_launch(void* const* d_in, const int* in_sizes, int n_in,
                              void* d_out, int out_size, void* d_ws, size_t ws_size,
                              hipStream_t stream) {
  (void)in_sizes; (void)n_in; (void)out_size; (void)ws_size;
  const float* x      = (const float*)d_in[0];   // [4,512,8192]
  const float* conv_w = (const float*)d_in[1];   // [512,1,5]
  const float* h_w    = (const float*)d_in[2];   // [512,512,1]
  const float* g_w    = (const float*)d_in[3];   // [510,512,1]
  float* out = (float*)d_out;                    // [4,512,8192] f32

  // workspace layout (bf16 halves): Wh(256K) Wg(256K) C(16.8M) H(16.8M) G(16.8M)
  unsigned short* Wh = (unsigned short*)d_ws;
  unsigned short* Wg = Wh + (size_t)O_ * D_;
  unsigned short* C  = Wg + (size_t)O_ * D_;
  unsigned short* H  = C  + (size_t)B_ * L_ * D_;
  unsigned short* G  = H  + (size_t)B_ * L_ * O_;

  prep_weights<<<2 * O_, 256, 0, stream>>>(h_w, g_w, Wh, Wg);
  dwconv<<<dim3(L_ / 128, D_ / 32, B_), 256, 0, stream>>>(x, conv_w, C);
  gemm_wmma<<<dim3(L_ / 128, O_ / 64, 2 * B_), 256, 0, stream>>>(Wh, Wg, C, H, G);
  scan_gru<<<dim3(O_, B_), 256, 0, stream>>>(H, G, out);
}
